// ScaledDotProductAttention_13159779795455
// MI455X (gfx1250) — matrix operand order repaired and checked by reading, not compiled, not run on hardware
//
#include <hip/hip_runtime.h>
#include <hip/hip_bf16.h>

typedef __attribute__((ext_vector_type(16))) __bf16 v16bf;
typedef __attribute__((ext_vector_type(8)))  float  v8f;
typedef __attribute__((ext_vector_type(4)))  float  f32x4;

#define B_DIM 32
#define S_DIM 2048
#define D_DIM 128
#define QT    16                  // queries per workgroup
#define SPAD  (S_DIM + 8)         // padded score row stride (floats)
#define VPADF (D_DIM + 4)         // padded V-stage row stride (floats, 16B-aligned)
#define INV_SCALE 0.08838834764831845f   // 1/sqrt(128)

__global__ __launch_bounds__(128)
void attn_fused_kernel(const float* __restrict__ Q,
                       const float* __restrict__ K,
                       const float* __restrict__ V,
                       float* __restrict__ Aout,
                       float* __restrict__ Out)
{
    __shared__ float sS[QT][SPAD];        // scores, then probabilities (fp32)
    __shared__ float sVf[2][32][VPADF];   // double-buffered V chunk (32 keys x 128 d)
    __shared__ float sMax[QT];
    __shared__ float sInv[QT];

    const int tid  = threadIdx.x;
    const int wave = tid >> 5;
    const int lane = tid & 31;
    const int b    = blockIdx.y;
    const int q0   = blockIdx.x * QT;

    const float* qbase = Q + ((size_t)b * S_DIM + q0) * D_DIM;
    const float* kbase = K + (size_t)b * S_DIM * D_DIM;
    const float* vbase = V + (size_t)b * S_DIM * D_DIM;

    // A-fragment lane mapping (16-bit A 16x32, same contiguous k order as B):
    // lanes 0-15 -> row M=lane, K=0..15; lanes 16-31 -> row M=lane-16, K=16..31
    const int arow  = lane & 15;
    const int ahalf = lane >> 4;

    // ---------------- Phase 1: S = (Q K^T) * inv_scale, causal mask -> LDS ----
    v16bf afrag[4];
    {
        const float* qrow = qbase + (size_t)arow * D_DIM;
        #pragma unroll
        for (int c = 0; c < 4; ++c) {
            #pragma unroll
            for (int j = 0; j < 8; ++j) {
                afrag[c][j]     = (__bf16)qrow[c * 32 + ahalf * 16 + j];
                afrag[c][8 + j] = (__bf16)qrow[c * 32 + ahalf * 16 + 8 + j];
            }
        }
    }

    // B-fragment lane mapping (16-bit B 32x16): lane n<16 -> col n, K=0..15;
    // lane n+16 -> col n, K=16..31 (contiguous run along D of K's row-major row)
    const int bcol  = lane & 15;
    const int bhalf = lane >> 4;

    const f32x4 ninf4 = {-__builtin_inff(), -__builtin_inff(),
                         -__builtin_inff(), -__builtin_inff()};

    for (int kt = wave; kt < S_DIM / 16; kt += 4) {
        const int k0 = kt * 16;
        if (k0 > q0 + QT - 1) {
            // fully-masked tile: skip GEMM, write -inf directly (16B stores)
            #pragma unroll
            for (int v4 = lane; v4 < 64; v4 += 32)
                *(f32x4*)&sS[v4 >> 2][k0 + ((v4 & 3) << 2)] = ninf4;
            continue;
        }
        v8f acc = {};
        #pragma unroll
        for (int c = 0; c < 4; ++c) {
            v16bf bfrag;
            const float* krow = kbase + (size_t)(k0 + bcol) * D_DIM + c * 32 + bhalf * 16;
            #pragma unroll
            for (int j = 0; j < 16; ++j) bfrag[j] = (__bf16)krow[j];
            acc = __builtin_amdgcn_wmma_f32_16x16x32_bf16(
                      false, afrag[c], false, bfrag, (short)0, acc, false, false);
        }
        // D tile: VGPR r -> (M=r, N=lane) lanes 0-15; (M=r+8, N=lane-16) lanes 16-31
        const int n    = lane & 15;
        const int mofs = (lane >> 4) * 8;
        #pragma unroll
        for (int r = 0; r < 8; ++r) {
            const int m = r + mofs;
            float s = acc[r] * INV_SCALE;
            if (k0 + n > q0 + m) s = -__builtin_inff();
            sS[m][k0 + n] = s;
        }
    }
    __syncthreads();

    // ---------------- Phase 2: fp32 row max & exp-sum (8 lanes per row) -------
    {
        const int row = tid >> 3;
        const int sub = tid & 7;
        const int lim = q0 + row;              // columns > lim are -inf: skip
        float mx = -__builtin_inff();
        for (int c = sub; c <= lim; c += 8) mx = fmaxf(mx, sS[row][c]);
        #pragma unroll
        for (int o = 1; o < 8; o <<= 1) mx = fmaxf(mx, __shfl_xor(mx, o, 32));
        float sum = 0.f;
        for (int c = sub; c <= lim; c += 8) sum += __expf(sS[row][c] - mx);
        #pragma unroll
        for (int o = 1; o < 8; o <<= 1) sum += __shfl_xor(sum, o, 32);
        if (sub == 0) { sMax[row] = mx; sInv[row] = 1.0f / sum; }
    }
    __syncthreads();

    const int kcmax = (q0 + QT - 1) >> 5;      // last key chunk with P != 0

    // Issue async global->LDS DMA (ASYNCcnt-tracked) for V chunk 0 now, so it
    // loads entirely under the shadow of the phase-3a streaming store of `a`.
    const uint64_t vb64 = (uint64_t)(uintptr_t)vbase;
    auto issue_vchunk = [&](int kc, int buf) {
        #pragma unroll
        for (int j = 0; j < 8; ++j) {
            const int id  = tid + j * 128;            // vec4 id, 1024 per chunk
            const int r   = id >> 5;                  // 32 vec4 per 128-wide row
            const int col = (id & 31) << 2;
            const uint32_t lds  = (uint32_t)(uintptr_t)&sVf[buf][r][col];
            const uint32_t goff = (uint32_t)(((kc * 32 + r) * D_DIM + col) * 4);
            asm volatile("global_load_async_to_lds_b128 %0, %1, %2"
                         :: "v"(lds), "v"(goff), "s"(vb64) : "memory");
        }
    };
    issue_vchunk(0, 0);

    // ---------------- Phase 3a: normalize; NT-hinted coalesced store of `a` ---
    // 512 MiB grid-wide, written once, never re-read: keep it out of L2 so the
    // L2-resident K/V working set (64 MiB) survives.
    {
        float* arow_out = Aout + ((size_t)b * S_DIM + q0) * S_DIM;
        for (int i = tid; i < QT * S_DIM / 4; i += 128) {
            const int row = i >> 9;                 // 512 vec4 per row
            const int c4  = (i & 511) << 2;
            f32x4 s = *(const f32x4*)&sS[row][c4];
            f32x4 p;
            #pragma unroll
            for (int e = 0; e < 4; ++e)
                p[e] = __expf(s[e] - sMax[row]) * sInv[row];   // -inf -> 0
            *(f32x4*)&sS[row][c4] = p;
            __builtin_nontemporal_store(p, (f32x4*)(arow_out + (size_t)row * S_DIM + c4));
        }
    }
    __syncthreads();

    // ---------------- Phase 3b: Out = P @ V, double-buffered async V ----------
    v8f oacc[2] = {{}, {}};
    const int d0w = wave * 32;                     // wave's 32 output dims

    for (int kc = 0; kc <= kcmax; ++kc) {
        const int buf = kc & 1;
        if (kc < kcmax) {
            issue_vchunk(kc + 1, buf ^ 1);         // prefetch next chunk
            // in-flight: 8 (current, older) + 8 (next); <=8 retires current
            asm volatile("s_wait_asynccnt 0x8" ::: "memory");
        } else {
            asm volatile("s_wait_asynccnt 0x0" ::: "memory");
        }
        __syncthreads();                           // current buffer visible to all

        // A-fragment: P 16x32 chunk from LDS (fp32 -> bf16)
        v16bf pfrag;
        {
            const float* prow = &sS[arow][kc * 32];
            #pragma unroll
            for (int j = 0; j < 8; ++j) {
                pfrag[j]     = (__bf16)prow[ahalf * 16 + j];
                pfrag[8 + j] = (__bf16)prow[ahalf * 16 + 8 + j];
            }
        }
        #pragma unroll
        for (int nt = 0; nt < 2; ++nt) {
            v16bf vfrag;
            const int d  = d0w + nt * 16 + (lane & 15);
            const int kb = (lane >> 4) * 16;
            #pragma unroll
            for (int j = 0; j < 16; ++j) vfrag[j] = (__bf16)sVf[buf][kb + j][d];
            oacc[nt] = __builtin_amdgcn_wmma_f32_16x16x32_bf16(
                           false, pfrag, false, vfrag, (short)0, oacc[nt], false, false);
        }
        __syncthreads();                           // done reading before overwrite
    }

    // write 16x32 output tile
    {
        float* obase = Out + ((size_t)b * S_DIM + q0) * D_DIM;
        const int n    = lane & 15;
        const int mofs = (lane >> 4) * 8;
        #pragma unroll
        for (int nt = 0; nt < 2; ++nt)
            #pragma unroll
            for (int r = 0; r < 8; ++r)
                obase[(size_t)(r + mofs) * D_DIM + d0w + nt * 16 + n] = oacc[nt][r];
    }
}

extern "C" void kernel_launch(void* const* d_in, const int* in_sizes, int n_in,
                              void* d_out, int out_size, void* d_ws, size_t ws_size,
                              hipStream_t stream) {
    const float* q = (const float*)d_in[0];
    const float* k = (const float*)d_in[1];
    const float* v = (const float*)d_in[2];
    // d_in[3] is the boolean causal mask (strictly j>i); regenerated analytically.
    float* aout = (float*)d_out;                                  // [B,S,S]
    float* out  = aout + (size_t)B_DIM * S_DIM * S_DIM;           // [B,S,D]
    dim3 grid(S_DIM / QT, B_DIM);
    attn_fused_kernel<<<grid, dim3(128), 0, stream>>>(q, k, v, aout, out);
}
